// MoE_39487929319966
// MI455X (gfx1250) — matrix operand order repaired and checked by reading, not compiled, not run on hardware
//
#include <hip/hip_runtime.h>

// MoE top-2 routed expert GEMM for MI455X (gfx1250, wave32, WMMA).
// Heavy path uses v_wmma_f32_16x16x32_bf16 with a hi/lo bf16 split
// (hi*hi + hi*lo + lo*hi) for near-fp32 accuracy at near-memory-roofline cost.

#define B_TOK 131072
#define DIN   512
#define DOUT  512
#define NE    16

typedef __attribute__((ext_vector_type(16))) __bf16 v16bf;
typedef __attribute__((ext_vector_type(8)))  float  v8f;

__device__ __forceinline__ unsigned short f32_bf16(float f) {
  unsigned u = __float_as_uint(f);
  u += 0x7fffu + ((u >> 16) & 1u);            // round-to-nearest-even
  return (unsigned short)(u >> 16);
}
__device__ __forceinline__ float bf16_f32(unsigned short h) {
  return __uint_as_float((unsigned)h << 16);
}

// ---------------------------------------------------------------- zero init
__global__ __launch_bounds__(256) void k_zero(float4* __restrict__ out4,
                                              size_t n4, int* __restrict__ counts) {
  size_t i = (size_t)blockIdx.x * blockDim.x + threadIdx.x;
  size_t stride = (size_t)gridDim.x * blockDim.x;
  float4 z; z.x = z.y = z.z = z.w = 0.0f;
  for (; i < n4; i += stride) out4[i] = z;
  if (blockIdx.x == 0 && threadIdx.x < NE) counts[threadIdx.x] = 0;
}

// ---------------------------------------------------------------- gating GEMM
// One thread per token: 16 dot products of length 512. x row loads are
// float4; the 2KB row stride is fine since each wave's 32 cachelines stay
// L0-resident across the 8 reuse iterations. Wg reads are wave-uniform
// (broadcast). Per-token norm reduced deterministically per block.
__global__ __launch_bounds__(256) void k_gating(const float* __restrict__ x,
                                                const float* __restrict__ Wg,
                                                const float* __restrict__ bg,
                                                float* __restrict__ logits,
                                                float* __restrict__ partials) {
  const int token = blockIdx.x * 256 + threadIdx.x;
  float acc[NE];
#pragma unroll
  for (int e = 0; e < NE; ++e) acc[e] = bg[e];
  const float* xr = x + (size_t)token * DIN;
  for (int k = 0; k < DIN; k += 4) {
    float xa[4];
    *(float4*)xa = *(const float4*)(xr + k);
#pragma unroll
    for (int j = 0; j < 4; ++j) {
      float wv[NE];
      const float4* wr = (const float4*)(Wg + (size_t)(k + j) * NE);
#pragma unroll
      for (int q = 0; q < 4; ++q) ((float4*)wv)[q] = wr[q];
      const float xs = xa[j];
#pragma unroll
      for (int e = 0; e < NE; ++e) acc[e] += xs * wv[e];
    }
  }
  float ss = 0.0f;
#pragma unroll
  for (int e = 0; e < NE; ++e) {
    logits[(size_t)token * NE + e] = acc[e];
    ss += acc[e] * acc[e];
  }
  const float gx = sqrtf(ss);

  __shared__ float red[256];
  red[threadIdx.x] = gx;
  __syncthreads();
  for (int s = 128; s > 0; s >>= 1) {
    if (threadIdx.x < s) red[threadIdx.x] += red[threadIdx.x + s];
    __syncthreads();
  }
  if (threadIdx.x == 0) partials[blockIdx.x] = red[0];
}

// ---------------------------------------------------------------- mean of Gx
__global__ __launch_bounds__(256) void k_mean(const float* __restrict__ partials,
                                              float* __restrict__ meanp) {
  __shared__ float red[256];
  const int t = threadIdx.x;
  red[t] = partials[t] + partials[t + 256];
  __syncthreads();
  for (int s = 128; s > 0; s >>= 1) {
    if (t < s) red[t] += red[t + s];
    __syncthreads();
  }
  if (t == 0) meanp[0] = red[0] / (float)B_TOK;
}

// ---------------------------------------------------------------- routing
__global__ __launch_bounds__(256) void k_route(const float* __restrict__ logits,
                                               const float* __restrict__ gamma,
                                               const float* __restrict__ beta,
                                               const float* __restrict__ meanp,
                                               int* __restrict__ counts,
                                               int* __restrict__ idx_list,
                                               float* __restrict__ w_list) {
  const int token = blockIdx.x * 256 + threadIdx.x;
  float l[NE];
  const float4* lr = (const float4*)(logits + (size_t)token * NE);
#pragma unroll
  for (int q = 0; q < 4; ++q) ((float4*)l)[q] = lr[q];

  float ss = 0.0f;
#pragma unroll
  for (int e = 0; e < NE; ++e) ss += l[e] * l[e];
  const float nx = sqrtf(ss) / (meanp[0] + 1e-6f);

#pragma unroll
  for (int e = 0; e < NE; ++e) l[e] = gamma[e] * (l[e] * nx) + beta[e] + l[e];

  float mx = l[0];
#pragma unroll
  for (int e = 1; e < NE; ++e) mx = fmaxf(mx, l[e]);
  float ef[NE];
  float se = 0.0f;
#pragma unroll
  for (int e = 0; e < NE; ++e) { ef[e] = expf(l[e] - mx); se += ef[e]; }
  const float inv = 1.0f / se;

  // top-1 (first occurrence on ties, like lax.top_k)
  int i1 = 0; float v1 = ef[0];
#pragma unroll
  for (int e = 1; e < NE; ++e) if (ef[e] > v1) { v1 = ef[e]; i1 = e; }
  // top-2
  int i2 = -1; float v2 = -1.0f;
#pragma unroll
  for (int e = 0; e < NE; ++e) if (e != i1 && ef[e] > v2) { v2 = ef[e]; i2 = e; }

  int s1 = atomicAdd(&counts[i1], 1);
  idx_list[(size_t)i1 * B_TOK + s1] = token;
  w_list [(size_t)i1 * B_TOK + s1] = v1 * inv;
  int s2 = atomicAdd(&counts[i2], 1);
  idx_list[(size_t)i2 * B_TOK + s2] = token;
  w_list [(size_t)i2 * B_TOK + s2] = v2 * inv;
}

// ---------------------------------------------------------------- expert GEMM
// Block: 256 threads (8 wave32s), tile 128(M) x 128(N), K-step 32, double-
// buffered LDS (one barrier per step). A rows gathered via per-expert token
// list; x and We are converted fp32 -> bf16 hi/lo and stored pre-swizzled in
// the CDNA5 16-bit A/B fragment layouts so each lane reads its v16bf frag as
// one contiguous 32B LDS access.
//   A frag (16x32 MxK): lane = (m&15) + ((k>>3)&1)*16 ; half = (k&7) + (k>>4)*8
//   B frag (32x16 KxN): lane = (n&15) + ((k>>3)&1)*16 ; half = (k&7) + (k>>4)*8
//   C frag (16x16):     lane(col) = n&15 (+16 dup) ; vgpr r -> m = r + 8*(lane>=16)
__global__ __launch_bounds__(256) void k_expert_gemm(const float* __restrict__ x,
                                                     const float* __restrict__ We,
                                                     const float* __restrict__ be,
                                                     const int*   __restrict__ counts,
                                                     const int*   __restrict__ idx_list,
                                                     const float* __restrict__ w_list,
                                                     float* __restrict__ out) {
  const int e   = blockIdx.z;
  const int cnt = counts[e];
  const int m0  = blockIdx.y * 128;
  if (m0 >= cnt) return;
  const int nblk = blockIdx.x;                    // 0..3 (128 cols each)

  __shared__ __align__(32) unsigned short As[2][2][4096];  // [buf][hi/lo][8 frags * 512 halves]
  __shared__ __align__(32) unsigned short Bs[2][2][4096];

  const int tid   = threadIdx.x;
  const int lane  = tid & 31;
  const int wave  = tid >> 5;
  const int waveM = wave & 3;                     // 4 waves along M (32 rows each)
  const int waveN = wave >> 2;                    // 2 waves along N (64 cols each)

  // ---- A-loader coords: 2 threads per row, 16 k each
  const int am     = tid >> 1;                    // 0..127
  const int akb    = (tid & 1) << 4;              // 0 or 16
  const int amfrag = am >> 4;
  const int amr    = am & 15;
  const int aslot  = m0 + am;
  const bool avalid = (aslot < cnt);
  const int atoken = avalid ? idx_list[(size_t)e * B_TOK + aslot] : 0;
  const float* axp = x + (size_t)atoken * DIN + akb;

  // ---- B-loader coords: 2 k-rows x 8 n-cols per thread
  const int bk  = (tid >> 4) << 1;                // 0,2,...,30
  const int bn0 = (tid & 15) << 3;                // 0..120
  const float* bwp = We + (size_t)e * DIN * DOUT + (size_t)bk * DOUT + nblk * 128 + bn0;

  v8f acc[2][4] = {};

  for (int k0 = 0, it = 0; k0 < DIN; k0 += 32, ++it) {
    const int buf = it & 1;

    // global -> regs
    float fa[16];
    if (avalid) {
#pragma unroll
      for (int i = 0; i < 4; ++i) ((float4*)fa)[i] = ((const float4*)(axp + k0))[i];
    } else {
#pragma unroll
      for (int i = 0; i < 16; ++i) fa[i] = 0.0f;
    }
    float fb0[8], fb1[8];
    {
      const float* p = bwp + (size_t)k0 * DOUT;
#pragma unroll
      for (int i = 0; i < 2; ++i) {
        ((float4*)fb0)[i] = ((const float4*)p)[i];
        ((float4*)fb1)[i] = ((const float4*)(p + DOUT))[i];
      }
    }

    // convert + swizzled LDS store (A)
    unsigned* Ah = (unsigned*)&As[buf][0][0];
    unsigned* Al = (unsigned*)&As[buf][1][0];
#pragma unroll
    for (int jp = 0; jp < 8; ++jp) {
      const int k = akb + jp * 2;                 // even
      const float f0 = fa[jp * 2], f1 = fa[jp * 2 + 1];
      const unsigned short h0 = f32_bf16(f0), h1 = f32_bf16(f1);
      const unsigned short l0 = f32_bf16(f0 - bf16_f32(h0));
      const unsigned short l1 = f32_bf16(f1 - bf16_f32(h1));
      const int ln  = amr + (((k >> 3) & 1) << 4);
      const int hh  = (k & 7) + ((k >> 4) << 3);  // even
      const int idx = amfrag * 256 + ln * 8 + (hh >> 1);
      Ah[idx] = (unsigned)h0 | ((unsigned)h1 << 16);
      Al[idx] = (unsigned)l0 | ((unsigned)l1 << 16);
    }
    // convert + swizzled LDS store (B): pack (k, k+1) along halves
    unsigned* Bh = (unsigned*)&Bs[buf][0][0];
    unsigned* Bl = (unsigned*)&Bs[buf][1][0];
#pragma unroll
    for (int n = 0; n < 8; ++n) {
      const int nn = bn0 + n;
      const float f0 = fb0[n], f1 = fb1[n];
      const unsigned short h0 = f32_bf16(f0), h1 = f32_bf16(f1);
      const unsigned short l0 = f32_bf16(f0 - bf16_f32(h0));
      const unsigned short l1 = f32_bf16(f1 - bf16_f32(h1));
      const int ln  = (nn & 15) + (((bk >> 3) & 1) << 4);
      const int hh  = (bk & 7) + ((bk >> 4) << 3); // even
      const int idx = (nn >> 4) * 256 + ln * 8 + (hh >> 1);
      Bh[idx] = (unsigned)h0 | ((unsigned)h1 << 16);
      Bl[idx] = (unsigned)l0 | ((unsigned)l1 << 16);
    }

    __syncthreads();

    // fragments + WMMA
    const v16bf* pAh = (const v16bf*)&As[buf][0][0];
    const v16bf* pAl = (const v16bf*)&As[buf][1][0];
    const v16bf* pBh = (const v16bf*)&Bs[buf][0][0];
    const v16bf* pBl = (const v16bf*)&Bs[buf][1][0];
    v16bf a_hi[2], a_lo[2], b_hi[4], b_lo[4];
#pragma unroll
    for (int mf = 0; mf < 2; ++mf) {
      const int tm = waveM * 2 + mf;
      a_hi[mf] = pAh[tm * 32 + lane];
      a_lo[mf] = pAl[tm * 32 + lane];
    }
#pragma unroll
    for (int nf = 0; nf < 4; ++nf) {
      const int tn = waveN * 4 + nf;
      b_hi[nf] = pBh[tn * 32 + lane];
      b_lo[nf] = pBl[tn * 32 + lane];
    }
#pragma unroll
    for (int mf = 0; mf < 2; ++mf) {
#pragma unroll
      for (int nf = 0; nf < 4; ++nf) {
        acc[mf][nf] = __builtin_amdgcn_wmma_f32_16x16x32_bf16(
            false, a_hi[mf], false, b_hi[nf], (short)0, acc[mf][nf], false, false);
        acc[mf][nf] = __builtin_amdgcn_wmma_f32_16x16x32_bf16(
            false, a_hi[mf], false, b_lo[nf], (short)0, acc[mf][nf], false, false);
        acc[mf][nf] = __builtin_amdgcn_wmma_f32_16x16x32_bf16(
            false, a_lo[mf], false, b_hi[nf], (short)0, acc[mf][nf], false, false);
      }
    }
  }

  // epilogue: weight, bias, scatter-accumulate (exactly 2 adds per out elem
  // from exact zero -> bitwise deterministic)
  const size_t eB = (size_t)e * B_TOK;
#pragma unroll
  for (int mf = 0; mf < 2; ++mf) {
    int   toks[8];
    float wv[8];
    const int mbase = m0 + waveM * 32 + mf * 16 + ((lane >> 4) << 3);
#pragma unroll
    for (int r = 0; r < 8; ++r) {
      const int slot = mbase + r;
      if (slot < cnt) { toks[r] = idx_list[eB + slot]; wv[r] = w_list[eB + slot]; }
      else            { toks[r] = -1;                  wv[r] = 0.0f; }
    }
#pragma unroll
    for (int nf = 0; nf < 4; ++nf) {
      const int nglob = nblk * 128 + (waveN * 4 + nf) * 16 + (lane & 15);
      const float bias = be[e * DOUT + nglob];
#pragma unroll
      for (int r = 0; r < 8; ++r) {
        if (toks[r] >= 0) {
          atomicAdd(out + (size_t)toks[r] * DOUT + nglob,
                    wv[r] * (acc[mf][nf][r] + bias));
        }
      }
    }
  }
}

// ---------------------------------------------------------------- launch
extern "C" void kernel_launch(void* const* d_in, const int* in_sizes, int n_in,
                              void* d_out, int out_size, void* d_ws, size_t ws_size,
                              hipStream_t stream) {
  (void)in_sizes; (void)n_in; (void)out_size; (void)ws_size;
  const float* x     = (const float*)d_in[0];
  const float* Wg    = (const float*)d_in[1];
  const float* bg    = (const float*)d_in[2];
  const float* gamma = (const float*)d_in[3];
  const float* beta  = (const float*)d_in[4];
  const float* We    = (const float*)d_in[5];
  const float* be    = (const float*)d_in[6];
  float* out = (float*)d_out;

  // workspace layout (~25 MB)
  float* ws_logits = (float*)d_ws;                          // B*16
  float* ws_part   = ws_logits + (size_t)B_TOK * NE;        // 512
  float* ws_mean   = ws_part + 512;                         // 1
  int*   ws_counts = (int*)(ws_mean + 1);                   // 16
  int*   ws_idx    = ws_counts + NE;                        // 16*B
  float* ws_w      = (float*)(ws_idx + (size_t)NE * B_TOK); // 16*B

  k_zero<<<2048, 256, 0, stream>>>((float4*)out, (size_t)B_TOK * DOUT / 4, ws_counts);
  k_gating<<<B_TOK / 256, 256, 0, stream>>>(x, Wg, bg, ws_logits, ws_part);
  k_mean<<<1, 256, 0, stream>>>(ws_part, ws_mean);
  k_route<<<B_TOK / 256, 256, 0, stream>>>(ws_logits, gamma, beta, ws_mean,
                                           ws_counts, ws_idx, ws_w);
  dim3 grid(DOUT / 128, B_TOK / 128, NE);
  k_expert_gemm<<<grid, 256, 0, stream>>>(x, We, be, ws_counts, ws_idx, ws_w, out);
}
